// attention_59030030516433
// MI455X (gfx1250) — matrix operand order repaired and checked by reading, not compiled, not run on hardware
//
#include <hip/hip_runtime.h>
#include <hip/hip_bf16.h>

// ---------- types for CDNA5 WMMA ----------
typedef __attribute__((ext_vector_type(16))) __bf16 bf16x16;
typedef __attribute__((ext_vector_type(8)))  float  v8f;

union FragU { unsigned int u[8]; bf16x16 v; };

__device__ __forceinline__ unsigned short f2bf(float f) {
  unsigned int u = __float_as_uint(f);
  unsigned int r = u + 0x7FFFu + ((u >> 16) & 1u);   // RNE
  return (unsigned short)(r >> 16);
}

// A-matrix 16x32 bf16 fragment (ISA 7.12.2): lane m=lane&15, g=lane>>4.
// v0..3 hold K = g*8 + {0..7}; v4..7 hold K = 16 + g*8 + {0..7}, packed pairs.
__device__ __forceinline__ bf16x16 ldsFragA(const unsigned short* s, int ld, int kbase, int lane) {
  const int m = lane & 15, g = lane >> 4;
  FragU f;
#pragma unroll
  for (int r = 0; r < 8; ++r) {
    const int kk = (r < 4) ? (g * 8 + 2 * r) : (16 + g * 8 + 2 * (r - 4));
    f.u[r] = *reinterpret_cast<const unsigned int*>(s + m * ld + kbase + kk);
  }
  return f.v;
}

// B-matrix 32x16 bf16 fragment: lane n=lane&15; v0..3 hold K = g*8 + {0..7}; v4..7 hold K = 16 + g*8 + {0..7}.
// LDS tile is stored [N][K] (i.e. W row-major for W^T operand).
__device__ __forceinline__ bf16x16 ldsFragB(const unsigned short* s, int ld, int kbase, int lane) {
  const int n = lane & 15, g = lane >> 4;
  FragU f;
#pragma unroll
  for (int r = 0; r < 8; ++r) {
    const int kk = (r < 4) ? (g * 8 + 2 * r) : (16 + g * 8 + 2 * (r - 4));
    f.u[r] = *reinterpret_cast<const unsigned int*>(s + n * ld + kbase + kk);
  }
  return f.v;
}

__device__ __forceinline__ v8f wmma_bf16(bf16x16 a, bf16x16 b, v8f c) {
  return __builtin_amdgcn_wmma_f32_16x16x32_bf16(false, a, false, b, (short)0, c, false, false);
}

// ---------- problem constants ----------
constexpr int  Bsz = 2, Nseq = 2048, Dm = 1024, Hh = 16, HD = 64;
constexpr long MROWS = (long)Bsz * Nseq;                 // 4096
constexpr long PART  = (long)Bsz * Hh * Nseq * HD;       // one of q/k/v, elems

// ---------- fp32 -> bf16 convert ----------
__global__ void cvt_f32_bf16(const float* __restrict__ in, unsigned short* __restrict__ out, int n) {
  int i = blockIdx.x * blockDim.x + threadIdx.x;
  if (i < n) out[i] = f2bf(in[i]);
}

// ---------- GEMM: C[M,N] = A[M,K] * W[N,K]^T + bias ----------
// Block: 256 threads (8 waves, 4x2), tile 128x128, BK=32.
constexpr int BM = 128, BN = 128, BKc = 32, LDT = 40;   // 40-elem padded LDS stride

template <bool QKV>
__global__ __launch_bounds__(256) void gemm_bf16_nt(const unsigned short* __restrict__ A,
                                                    const unsigned short* __restrict__ W,
                                                    const float* __restrict__ bias,
                                                    unsigned short* __restrict__ qkv_out,
                                                    float* __restrict__ f_out, int K) {
  __shared__ __align__(16) unsigned short As[BM * LDT];
  __shared__ __align__(16) unsigned short Bs[BN * LDT];
  const int tid = threadIdx.x, lane = tid & 31, wave = tid >> 5;
  const int wm = wave & 3, wn = wave >> 2;
  const int m0 = blockIdx.y * BM, n0 = blockIdx.x * BN;

  v8f zero = {};
  v8f c[2][4];
#pragma unroll
  for (int i = 0; i < 2; ++i)
#pragma unroll
    for (int j = 0; j < 4; ++j) c[i][j] = zero;

  const int lrow = tid >> 1, lseg = tid & 1;
  const uint4* Ag = reinterpret_cast<const uint4*>(A + (size_t)(m0 + lrow) * K) + lseg * 2;
  const uint4* Wg = reinterpret_cast<const uint4*>(W + (size_t)(n0 + lrow) * K) + lseg * 2;
  uint4* Ad = reinterpret_cast<uint4*>(&As[lrow * LDT + lseg * 16]);
  uint4* Bd = reinterpret_cast<uint4*>(&Bs[lrow * LDT + lseg * 16]);

  for (int k0 = 0; k0 < K; k0 += BKc) {
    const uint4* ag = Ag + (k0 >> 3);
    const uint4* wg = Wg + (k0 >> 3);
    uint4 a0 = ag[0], a1 = ag[1];
    uint4 b0 = wg[0], b1 = wg[1];
    __builtin_prefetch((const void*)(ag + 4), 0, 1);
    __builtin_prefetch((const void*)(wg + 4), 0, 1);
    Ad[0] = a0; Ad[1] = a1;
    Bd[0] = b0; Bd[1] = b1;
    __syncthreads();

    bf16x16 af[2];
#pragma unroll
    for (int i = 0; i < 2; ++i) af[i] = ldsFragA(&As[(wm * 32 + i * 16) * LDT], LDT, 0, lane);
#pragma unroll
    for (int j = 0; j < 4; ++j) {
      bf16x16 bf = ldsFragB(&Bs[(wn * 64 + j * 16) * LDT], LDT, 0, lane);
#pragma unroll
      for (int i = 0; i < 2; ++i) c[i][j] = wmma_bf16(af[i], bf, c[i][j]);
    }
    __syncthreads();
  }

  // Epilogue. C layout: lane col = lane&15, slot r -> row r + 8*(lane>>4).
  const int g = lane >> 4, nn = lane & 15;
#pragma unroll
  for (int i = 0; i < 2; ++i)
#pragma unroll
    for (int j = 0; j < 4; ++j) {
      const int col = n0 + wn * 64 + j * 16 + nn;
      const float bv = bias[col];
#pragma unroll
      for (int r = 0; r < 8; ++r) {
        const int row = m0 + wm * 32 + i * 16 + r + 8 * g;
        const float val = c[i][j][r] + bv;
        if constexpr (QKV) {
          const int which = col >> 10, d = col & 1023;
          const int hh = d >> 6, hd = d & 63;
          const int bb = row >> 11, n = row & 2047;
          qkv_out[(size_t)which * PART + ((((size_t)bb * Hh + hh) * Nseq + n) << 6) + hd] = f2bf(val);
        } else {
          f_out[(size_t)row * Dm + col] = val;
        }
      }
    }
}

// ---------- flash attention ----------
// One block = (b, h, 128 q-rows). 8 waves x 16 q-rows. KV tiles of 64.
__global__ __launch_bounds__(256) void attn_kernel(const unsigned short* __restrict__ qkv,
                                                   unsigned short* __restrict__ attn_out) {
  constexpr int LD = 72;
  __shared__ __align__(16) unsigned short Qs[128 * LD];
  __shared__ __align__(16) unsigned short Ks[64 * LD];
  __shared__ __align__(16) unsigned short Vts[64 * LD];   // transposed: [d][kvrow]
  __shared__ __align__(16) unsigned short Ps[128 * LD];
  const int tid = threadIdx.x, lane = tid & 31, wave = tid >> 5;
  const int qc = blockIdx.x & 15;
  const int h  = (blockIdx.x >> 4) & 15;
  const int b  = blockIdx.x >> 8;
  const unsigned short* Qg = qkv + ((((size_t)b * Hh + h) * Nseq) << 6);
  const unsigned short* Kg = Qg + PART;
  const unsigned short* Vg = Qg + 2 * PART;
  const int qbase = qc * 128;

  { // stage Q tile 128x64
    const int row = tid >> 1, half = tid & 1;
    const uint4* src = reinterpret_cast<const uint4*>(Qg + ((size_t)(qbase + row) << 6) + half * 32);
    uint2* d2 = reinterpret_cast<uint2*>(&Qs[row * LD + half * 32]);
    uint4 v0 = src[0], v1 = src[1], v2 = src[2], v3 = src[3];
    d2[0] = make_uint2(v0.x, v0.y); d2[1] = make_uint2(v0.z, v0.w);
    d2[2] = make_uint2(v1.x, v1.y); d2[3] = make_uint2(v1.z, v1.w);
    d2[4] = make_uint2(v2.x, v2.y); d2[5] = make_uint2(v2.z, v2.w);
    d2[6] = make_uint2(v3.x, v3.y); d2[7] = make_uint2(v3.z, v3.w);
  }

  v8f zero = {};
  v8f ot[4];
#pragma unroll
  for (int j = 0; j < 4; ++j) ot[j] = zero;
  float mrun[8], lrun[8];
#pragma unroll
  for (int r = 0; r < 8; ++r) { mrun[r] = -1e30f; lrun[r] = 0.f; }
  const float scale = 0.125f;  // 1/sqrt(64)
  const int krow = tid >> 2, kseg = tid & 3;

  for (int kt = 0; kt < Nseq / 64; ++kt) {
    { // stage K tile (natural) and V tile (transposed)
      const uint4* src = reinterpret_cast<const uint4*>(Kg + ((size_t)(kt * 64 + krow) << 6) + kseg * 16);
      uint4 k0 = src[0], k1 = src[1];
      uint2* d2 = reinterpret_cast<uint2*>(&Ks[krow * LD + kseg * 16]);
      d2[0] = make_uint2(k0.x, k0.y); d2[1] = make_uint2(k0.z, k0.w);
      d2[2] = make_uint2(k1.x, k1.y); d2[3] = make_uint2(k1.z, k1.w);
      const uint4* vsrc = reinterpret_cast<const uint4*>(Vg + ((size_t)(kt * 64 + krow) << 6) + kseg * 16);
      union { uint4 q[2]; unsigned short s[16]; } vb;
      vb.q[0] = vsrc[0]; vb.q[1] = vsrc[1];
#pragma unroll
      for (int e = 0; e < 16; ++e) Vts[(kseg * 16 + e) * LD + krow] = vb.s[e];
    }
    __syncthreads();

    // S = Q K^T  (per wave: 16x64 = 4 tiles, 2 k-steps)
    v8f st[4];
#pragma unroll
    for (int j = 0; j < 4; ++j) st[j] = zero;
#pragma unroll
    for (int ks = 0; ks < 2; ++ks) {
      bf16x16 aq = ldsFragA(&Qs[(wave * 16) * LD], LD, ks * 32, lane);
#pragma unroll
      for (int j = 0; j < 4; ++j) {
        bf16x16 bk = ldsFragB(&Ks[(j * 16) * LD], LD, ks * 32, lane);
        st[j] = wmma_bf16(aq, bk, st[j]);
      }
    }
#pragma unroll
    for (int j = 0; j < 4; ++j) st[j] = st[j] * scale;

    // online softmax; slot r <-> q-row (r + 8*(lane>>4)); 16-lane-half reduce
#pragma unroll
    for (int r = 0; r < 8; ++r) {
      float mx = -1e30f;
#pragma unroll
      for (int j = 0; j < 4; ++j) mx = fmaxf(mx, st[j][r]);
#pragma unroll
      for (int off = 1; off < 16; off <<= 1) mx = fmaxf(mx, __shfl_xor(mx, off, 32));
      const float mnew = fmaxf(mrun[r], mx);
      const float corr = __expf(mrun[r] - mnew);
      float rsum = 0.f;
#pragma unroll
      for (int j = 0; j < 4; ++j) {
        float pv = __expf(st[j][r] - mnew);
        st[j][r] = pv;
        rsum += pv;
      }
#pragma unroll
      for (int off = 1; off < 16; off <<= 1) rsum += __shfl_xor(rsum, off, 32);
      mrun[r] = mnew;
      lrun[r] = lrun[r] * corr + rsum;
#pragma unroll
      for (int j = 0; j < 4; ++j) ot[j][r] *= corr;
    }

    // round-trip P through LDS: C layout -> A layout (per-wave private region)
    {
      const int g = lane >> 4, nn = lane & 15;
#pragma unroll
      for (int j = 0; j < 4; ++j)
#pragma unroll
        for (int r = 0; r < 8; ++r)
          Ps[(wave * 16 + r + 8 * g) * LD + j * 16 + nn] = f2bf(st[j][r]);
    }

    // O += P V
#pragma unroll
    for (int ks = 0; ks < 2; ++ks) {
      bf16x16 ap = ldsFragA(&Ps[(wave * 16) * LD], LD, ks * 32, lane);
#pragma unroll
      for (int j = 0; j < 4; ++j) {
        bf16x16 bv = ldsFragB(&Vts[(j * 16) * LD], LD, ks * 32, lane);
        ot[j] = wmma_bf16(ap, bv, ot[j]);
      }
    }
    __syncthreads();
  }

  // normalize + write [B,N,D] bf16 (head-interleaved so proj GEMM reads it flat)
  const int g = lane >> 4, nn = lane & 15;
#pragma unroll
  for (int r = 0; r < 8; ++r) {
    const float inv = 1.f / lrun[r];
    const int row = qbase + wave * 16 + r + 8 * g;
#pragma unroll
    for (int j = 0; j < 4; ++j)
      attn_out[(((size_t)b * Nseq + row) << 10) + h * 64 + j * 16 + nn] = f2bf(ot[j][r] * inv);
  }
}

// ---------- host launcher ----------
extern "C" void kernel_launch(void* const* d_in, const int* in_sizes, int n_in,
                              void* d_out, int out_size, void* d_ws, size_t ws_size,
                              hipStream_t stream) {
  (void)in_sizes; (void)n_in; (void)out_size; (void)ws_size;
  const float* x      = (const float*)d_in[0];
  const float* qkv_w  = (const float*)d_in[1];
  const float* qkv_b  = (const float*)d_in[2];
  const float* proj_w = (const float*)d_in[3];
  const float* proj_b = (const float*)d_in[4];
  float* out = (float*)d_out;

  unsigned short* p = (unsigned short*)d_ws;
  unsigned short* xb    = p; p += MROWS * Dm;          // x in bf16
  unsigned short* wqkv  = p; p += (size_t)3 * Dm * Dm; // qkv_w bf16
  unsigned short* wproj = p; p += (size_t)Dm * Dm;     // proj_w bf16
  unsigned short* qkvp  = p; p += 3 * PART;            // q,k,v per-head bf16
  unsigned short* attnb = p; p += MROWS * Dm;          // attention out bf16

  const int nx = (int)(MROWS * Dm), nwq = 3 * Dm * Dm, nwp = Dm * Dm;
  cvt_f32_bf16<<<nx / 256, 256, 0, stream>>>(x, xb, nx);
  cvt_f32_bf16<<<nwq / 256, 256, 0, stream>>>(qkv_w, wqkv, nwq);
  cvt_f32_bf16<<<nwp / 256, 256, 0, stream>>>(proj_w, wproj, nwp);

  gemm_bf16_nt<true><<<dim3(3 * Dm / BN, MROWS / BM), 256, 0, stream>>>(
      xb, wqkv, qkv_b, qkvp, nullptr, Dm);

  attn_kernel<<<Bsz * Hh * (Nseq / 128), 256, 0, stream>>>(qkvp, attnb);

  gemm_bf16_nt<false><<<dim3(Dm / BN, MROWS / BM), 256, 0, stream>>>(
      attnb, wproj, proj_b, nullptr, out, Dm);
}
